// NeuralODE_70025146794764
// MI455X (gfx1250) — matrix operand order repaired and checked by reading, not compiled, not run on hardware
//
#include <hip/hip_runtime.h>

// ---------------------------------------------------------------------------
// Fused Tsit5 Neural-ODE integrator for MI455X (gfx1250, wave32, WMMA).
// One wave owns 16 batch rows for the entire integration (no inter-wave deps
// after weight staging). f16 WMMA 16x16x32, f32 accumulate.
// Round 4: TDM (tensor_load_to_lds) staging of f16 weights from d_ws.
// ---------------------------------------------------------------------------

typedef __attribute__((ext_vector_type(16))) _Float16 v16h;
typedef __attribute__((ext_vector_type(8)))  _Float16 v8h;
typedef __attribute__((ext_vector_type(8)))  float    v8f;
typedef __attribute__((ext_vector_type(4)))  unsigned v4u;
typedef __attribute__((ext_vector_type(8)))  int      v8i;
typedef __attribute__((ext_vector_type(4)))  int      v4i;

#define AS3 __attribute__((address_space(3)))
typedef AS3 _Float16 lh;   // f16 in LDS
typedef AS3 float    lf;   // f32 in LDS

#define NWAVES  4      // waves per workgroup (share one LDS weight copy)
#define DIM     32     // ODE state dim
#define HID     256    // MLP hidden width
#define BATCH   2048
#define TSAVE   32
#define NSUB    2
#define NTHREADS (NWAVES * 32)

// Tsit5 coefficients
#define A21f 0.161f
#define A31f -0.008480655492356989f
#define A32f 0.335480655492357f
#define A41f 2.8971530571054935f
#define A42f -6.359448489975075f
#define A43f 4.3622954328695815f
#define A51f 5.325864828439257f
#define A52f -11.748883564062828f
#define A53f 7.4955393428898365f
#define A54f -0.09249506636175525f
#define A61f 5.86145544294642f
#define A62f -12.92096931784711f
#define A63f 8.159367898576159f
#define A64f -0.071584973281401f
#define A65f -0.028269050394068383f
#define B1f 0.09646076681806523f
#define B2f 0.01f
#define B3f 0.4798896504144996f
#define B4f 1.379008574103742f
#define B5f -3.290069515436081f
#define B6f 2.324710524099774f

// LDS byte offsets (all 16B aligned)
#define OFF_W1   0u                        // 256*256 f16 = 131072
#define OFF_W0   131072u                   // 256*32  f16 = 16384
#define OFF_W2   147456u                   // 32*256  f16 = 16384
#define OFF_B0   163840u                   // 256 f32
#define OFF_B1   164864u                   // 256 f32
#define OFF_B2   165888u                   // 32 f32
#define OFF_Y    166016u                   // NWAVES*16*32  f16
#define OFF_H1   (OFF_Y  + NWAVES*1024u)   // NWAVES*16*256 f16
#define OFF_H2   (OFF_H1 + NWAVES*8192u)
#define SMEM_BYTES (OFF_H2 + NWAVES*8192u) // = 235648

// f16 weight blob layout inside d_ws (element offsets)
#define WS_W1_OFF  0        // 65536 halves
#define WS_W0_OFF  65536    // 8192 halves
#define WS_W2_OFF  73728    // 8192 halves
#define WS_TOTAL_H 81920
#define WS_BYTES_NEEDED (WS_TOTAL_H * 2)

#if defined(__has_builtin)
#if __has_builtin(__builtin_amdgcn_tensor_load_to_lds) && \
    __has_builtin(__builtin_amdgcn_s_wait_tensorcnt)
#define HAVE_TDM 1
#endif
#endif

// softplus(x) = max(x,0) + ln(1 + e^{-|x|})
// Raw v_exp_f32 / v_log_f32: log argument is in (1,2] (never denormal, never
// overflowing) and exp2 underflow-to-zero is exact, so libm fixups are dead.
__device__ __forceinline__ float softplus_f(float x) {
#if __has_builtin(__builtin_amdgcn_exp2f) && __has_builtin(__builtin_amdgcn_logf)
    float t = __builtin_amdgcn_exp2f(fabsf(x) * -1.4426950408889634f); // e^{-|x|}
    float l = __builtin_amdgcn_logf(1.0f + t);                         // log2(1+t)
    return fmaf(l, 0.6931471805599453f, fmaxf(x, 0.0f));
#else
    return fmaxf(x, 0.0f) + __logf(1.0f + __expf(-fabsf(x)));
#endif
}

__device__ __forceinline__ v8f wmma_f16(v16h a, v16h b, v8f c) {
    // D = A(16x32 f16) * B(32x16 f16) + C(16x16 f32)
    return __builtin_amdgcn_wmma_f32_16x16x32_f16(
        /*neg_a=*/false, a, /*neg_b=*/false, b,
        /*c_mod=*/(short)0, c, /*reuse_a=*/false, /*reuse_b=*/false);
}

#ifdef HAVE_TDM
// TDM: DMA a dense 2-D f16 tile (dim1 rows x dim0 cols, row-major, stride =
// dim0) from global memory into LDS. Descriptor per CDNA5 ISA:
// group0: count=1 | lds_addr | global_addr | type=2 ("image")
// group1: data_size=1 (2B), tensor_dim0/1, tile_dim0/1 = whole tensor,
//         tensor_dim0_stride = dim0; no multicast / barrier / pad / iterate.
__device__ __forceinline__ void tdm_load_2d_f16(unsigned lds_addr,
                                                const _Float16* gptr,
                                                unsigned dim0, unsigned dim1) {
    unsigned long long ga = (unsigned long long)(size_t)gptr;
    v4u g0;
    g0.x = 1u;                                            // count=1, user D#
    g0.y = lds_addr;                                      // LDS byte address
    g0.z = (unsigned)(ga & 0xFFFFFFFFull);                // global_addr[31:0]
    g0.w = (unsigned)((ga >> 32) & 0x1FFFFFFull) | (2u << 30); // [56:32] | type=2
    v8i g1;
    g1[0] = (int)(1u << 16);                              // wg_mask=0, data_size=2B
    g1[1] = (int)((dim0 & 0xFFFFu) << 16);                // tensor_dim0[15:0]
    g1[2] = (int)((dim0 >> 16) | ((dim1 & 0xFFFFu) << 16)); // dim0[31:16]|dim1[15:0]
    g1[3] = (int)((dim1 >> 16) | ((dim0 & 0xFFFFu) << 16)); // dim1[31:16]|tile_dim0
    g1[4] = (int)(dim1 & 0xFFFFu);                        // tile_dim1 | tile_dim2=0
    g1[5] = (int)dim0;                                    // tensor_dim0_stride[31:0]
    g1[6] = 0;                                            // stride0[47:32]|stride1[15:0]
    g1[7] = 0;                                            // stride1[47:16]
    v4i gz = {0, 0, 0, 0};
#if __clang_major__ >= 23
    v8i gz8 = {0, 0, 0, 0, 0, 0, 0, 0};
    __builtin_amdgcn_tensor_load_to_lds(g0, g1, gz, gz, gz8, 0);
#else
    __builtin_amdgcn_tensor_load_to_lds(g0, g1, gz, gz, 0);
#endif
}
#endif

__global__ void convert_weights_kernel(const float* __restrict__ gW0,
                                       const float* __restrict__ gW1,
                                       const float* __restrict__ gW2,
                                       _Float16* __restrict__ dst) {
    int i = blockIdx.x * blockDim.x + threadIdx.x;
    if (i < 65536)      dst[WS_W1_OFF + i]         = (_Float16)gW1[i];
    else if (i < 73728) dst[WS_W0_OFF + i - 65536] = (_Float16)gW0[i - 65536];
    else if (i < WS_TOTAL_H) dst[WS_W2_OFF + i - 73728] = (_Float16)gW2[i - 73728];
}

__global__ __launch_bounds__(NTHREADS, 1)
void tsit5_node_kernel(const float* __restrict__ ts,  const float* __restrict__ y0,
                       const float* __restrict__ gW0, const float* __restrict__ gb0,
                       const float* __restrict__ gW1, const float* __restrict__ gb1,
                       const float* __restrict__ gW2, const float* __restrict__ gb2,
                       const _Float16* __restrict__ wsrc, int use_tdm,
                       float* __restrict__ out)
{
    extern __shared__ char smem[];
    // Low 32 bits of a generic LDS address == LDS byte offset (ISA flat->LDS
    // mapping). Build explicit addrspace(3) pointers so all on-chip traffic
    // lowers to ds_load/ds_store instead of flat_*.
    const unsigned sbase = (unsigned)(size_t)smem;
    lh* w1  = (lh*)(sbase + OFF_W1);
    lh* w0  = (lh*)(sbase + OFF_W0);
    lh* w2  = (lh*)(sbase + OFF_W2);
    lf* sb0 = (lf*)(sbase + OFF_B0);
    lf* sb1 = (lf*)(sbase + OFF_B1);
    lf* sb2 = (lf*)(sbase + OFF_B2);

    const int tid  = threadIdx.x;
    const int lane = tid & 31;
    const int wave = tid >> 5;

    // ---- biases: fp32 global -> f32 LDS (tiny, always direct) ----
    for (int i = tid; i < HID; i += NTHREADS) { sb0[i] = gb0[i]; sb1[i] = gb1[i]; }
    for (int i = tid; i < DIM; i += NTHREADS) { sb2[i] = gb2[i]; }

    // ---- weights: f16 into LDS, via Tensor Data Mover when available ----
#ifdef HAVE_TDM
    if (use_tdm) {
        if (wave == 0) {   // one wave drives the TDM (EXEC is ignored by TDM)
            tdm_load_2d_f16(sbase + OFF_W1, wsrc + WS_W1_OFF, HID, HID);
            tdm_load_2d_f16(sbase + OFF_W0, wsrc + WS_W0_OFF, DIM, HID);
            tdm_load_2d_f16(sbase + OFF_W2, wsrc + WS_W2_OFF, HID, DIM);
            __builtin_amdgcn_s_wait_tensorcnt(0);
        }
    } else
#endif
    {
        (void)wsrc;
        for (int i = tid; i < HID * HID; i += NTHREADS) w1[i] = (_Float16)gW1[i];
        for (int i = tid; i < HID * DIM; i += NTHREADS) w0[i] = (_Float16)gW0[i];
        for (int i = tid; i < DIM * HID; i += NTHREADS) w2[i] = (_Float16)gW2[i];
    }
    __syncthreads();
    // After this point each wave is fully independent: no more barriers.

    lh* ybuf = (lh*)(sbase + OFF_Y)  + wave * 16 * DIM;
    lh* h1   = (lh*)(sbase + OFF_H1) + wave * 16 * HID;
    lh* h2   = (lh*)(sbase + OFF_H2) + wave * 16 * HID;

    const int mlo = lane & 15;     // A/B row-lane; C column
    const int hih = lane >> 4;     // half-select
    const int rowbase = (blockIdx.x * NWAVES + wave) * 16;  // global batch row

    // A fragment (16xK f16, K-slice at kb) from row-major LDS [16][rowstride]
    auto frag_a = [&](const lh* base, int rowstride, int kb) -> v16h {
        const lh* p = base + mlo * rowstride + kb + hih * 8;
        v8h lo = *(const AS3 v8h*)(p);
        v8h hi = *(const AS3 v8h*)(p + 16);
        return __builtin_shufflevector(lo, hi, 0,1,2,3,4,5,6,7,8,9,10,11,12,13,14,15);
    };
    // B fragment (32x16, K-slice kb, N-tile nt) from row-major weight W[n][k]
    auto frag_b = [&](const lh* wbase, int rowstride, int nt, int kb) -> v16h {
        const lh* p = wbase + (nt * 16 + mlo) * rowstride + kb + hih * 8;
        v8h lo = *(const AS3 v8h*)(p);
        v8h hi = *(const AS3 v8h*)(p + 16);
        return __builtin_shufflevector(lo, hi, 0,1,2,3,4,5,6,7,8,9,10,11,12,13,14,15);
    };

    // write Tsit5 stage input (C-layout regs) to ybuf as f16
    auto stage_store = [&](const float* yt) {
        #pragma unroll
        for (int i = 0; i < 16; ++i) {
            int r = i & 7, nt = i >> 3;
            ybuf[(r + 8 * hih) * DIM + nt * 16 + mlo] = (_Float16)yt[i];
        }
    };

    // k = f(ybuf) : softplus(softplus(y W0^T + b0) W1^T + b1) W2^T + b2
    auto f_eval = [&](float* kout) {
        // ---- layer 1: [16x32] x [32x256] ----
        v16h ay = frag_a(ybuf, DIM, 0);
        #pragma unroll
        for (int nt = 0; nt < 16; ++nt) {
            v8f acc = {};
            acc = wmma_f16(ay, frag_b(w0, DIM, nt, 0), acc);
            float bias = sb0[nt * 16 + mlo];
            #pragma unroll
            for (int r = 0; r < 8; ++r)
                h1[(r + 8 * hih) * HID + nt * 16 + mlo] = (_Float16)softplus_f(acc[r] + bias);
        }
        // ---- layer 2: [16x256] x [256x256] ----
        v16h a2[8];
        #pragma unroll
        for (int kt = 0; kt < 8; ++kt) a2[kt] = frag_a(h1, HID, kt * 32);
        #pragma unroll
        for (int nt = 0; nt < 16; ++nt) {
            v8f acc = {};
            #pragma unroll
            for (int kt = 0; kt < 8; ++kt)
                acc = wmma_f16(a2[kt], frag_b(w1, HID, nt, kt * 32), acc);
            float bias = sb1[nt * 16 + mlo];
            #pragma unroll
            for (int r = 0; r < 8; ++r)
                h2[(r + 8 * hih) * HID + nt * 16 + mlo] = (_Float16)softplus_f(acc[r] + bias);
        }
        // ---- layer 3: [16x256] x [256x32] (linear) ----
        v16h a3[8];
        #pragma unroll
        for (int kt = 0; kt < 8; ++kt) a3[kt] = frag_a(h2, HID, kt * 32);
        #pragma unroll
        for (int nt = 0; nt < 2; ++nt) {
            v8f acc = {};
            #pragma unroll
            for (int kt = 0; kt < 8; ++kt)
                acc = wmma_f16(a3[kt], frag_b(w2, HID, nt, kt * 32), acc);
            float bias = sb2[nt * 16 + mlo];
            #pragma unroll
            for (int r = 0; r < 8; ++r) kout[nt * 8 + r] = acc[r] + bias;
        }
    };

    // ---- load y0 into C-layout regs; emit save point t=0 ----
    float y[16];
    #pragma unroll
    for (int i = 0; i < 16; ++i) {
        int r = i & 7, nt = i >> 3;
        int m = rowbase + r + 8 * hih;
        int n = nt * 16 + mlo;
        y[i] = y0[m * DIM + n];
        out[(size_t)m * DIM + n] = y[i];   // out[0][m][n]
    }

    float k1[16], k2[16], k3[16], k4[16], k5[16], k6[16], yt[16];

    for (int t = 0; t < TSAVE - 1; ++t) {
        const float h = (ts[t + 1] - ts[t]) * (1.0f / NSUB);
        for (int sub = 0; sub < NSUB; ++sub) {
            stage_store(y);  f_eval(k1);
            #pragma unroll
            for (int i = 0; i < 16; ++i) yt[i] = y[i] + h * (A21f * k1[i]);
            stage_store(yt); f_eval(k2);
            #pragma unroll
            for (int i = 0; i < 16; ++i) yt[i] = y[i] + h * (A31f * k1[i] + A32f * k2[i]);
            stage_store(yt); f_eval(k3);
            #pragma unroll
            for (int i = 0; i < 16; ++i) yt[i] = y[i] + h * (A41f * k1[i] + A42f * k2[i] + A43f * k3[i]);
            stage_store(yt); f_eval(k4);
            #pragma unroll
            for (int i = 0; i < 16; ++i) yt[i] = y[i] + h * (A51f * k1[i] + A52f * k2[i] + A53f * k3[i] + A54f * k4[i]);
            stage_store(yt); f_eval(k5);
            #pragma unroll
            for (int i = 0; i < 16; ++i) yt[i] = y[i] + h * (A61f * k1[i] + A62f * k2[i] + A63f * k3[i] + A64f * k4[i] + A65f * k5[i]);
            stage_store(yt); f_eval(k6);
            #pragma unroll
            for (int i = 0; i < 16; ++i)
                y[i] += h * (B1f * k1[i] + B2f * k2[i] + B3f * k3[i] + B4f * k4[i] + B5f * k5[i] + B6f * k6[i]);
        }
        // save point t+1
        #pragma unroll
        for (int i = 0; i < 16; ++i) {
            int r = i & 7, nt = i >> 3;
            int m = rowbase + r + 8 * hih;
            out[((size_t)(t + 1) * BATCH + m) * DIM + nt * 16 + mlo] = y[i];
        }
    }
}

extern "C" void kernel_launch(void* const* d_in, const int* in_sizes, int n_in,
                              void* d_out, int out_size, void* d_ws, size_t ws_size,
                              hipStream_t stream) {
    (void)in_sizes; (void)n_in; (void)out_size;
    const float* ts  = (const float*)d_in[0];
    const float* y0  = (const float*)d_in[1];
    const float* W0  = (const float*)d_in[2];
    const float* b0  = (const float*)d_in[3];
    const float* W1  = (const float*)d_in[4];
    const float* b1  = (const float*)d_in[5];
    const float* W2  = (const float*)d_in[6];
    const float* b2  = (const float*)d_in[7];
    float* out = (float*)d_out;

    _Float16* wsrc = (_Float16*)d_ws;
    const int use_tdm = (d_ws != nullptr && ws_size >= (size_t)WS_BYTES_NEEDED) ? 1 : 0;

    if (use_tdm) {
        // Pre-convert weights fp32 -> f16 into the workspace blob (TDM source).
        convert_weights_kernel<<<(WS_TOTAL_H + 255) / 256, 256, 0, stream>>>(
            W0, W1, W2, wsrc);
    }

    // Host-side attribute set (not a stream op; safe under graph capture).
    hipFuncSetAttribute(reinterpret_cast<const void*>(tsit5_node_kernel),
                        hipFuncAttributeMaxDynamicSharedMemorySize, (int)SMEM_BYTES);

    dim3 grid(BATCH / (16 * NWAVES));   // 32 workgroups
    dim3 block(NTHREADS);               // 4 waves
    hipLaunchKernelGGL(tsit5_node_kernel, grid, block, SMEM_BYTES, stream,
                       ts, y0, W0, b0, W1, b1, W2, b2, wsrc, use_tdm, out);
}
